// DiTBlock_58119497449706
// MI455X (gfx1250) — matrix operand order repaired and checked by reading, not compiled, not run on hardware
//
#include <hip/hip_runtime.h>
#include <hip/hip_bf16.h>
#include <math.h>

// ---------------------------------------------------------------------------
// Types for CDNA5 WMMA (wave32, gfx1250)
// ---------------------------------------------------------------------------
typedef __bf16 bf16_t;
typedef __attribute__((ext_vector_type(16))) __bf16 bf16x16;
typedef __attribute__((ext_vector_type(8)))  __bf16 bf16x8;
typedef __attribute__((ext_vector_type(8)))  float  f32x8;

__device__ __forceinline__ f32x8 wmma_bf16(bf16x16 a, bf16x16 b, f32x8 c) {
    // v_wmma_f32_16x16x32_bf16: D = A(16x32) * B(32x16) + C(16x16)
    return __builtin_amdgcn_wmma_f32_16x16x32_bf16(
        /*neg_a=*/false, a, /*neg_b=*/false, b,
        /*c_mod=*/(short)0, c, /*reuse_a=*/false, /*reuse_b=*/false);
}

__device__ __forceinline__ bf16x16 cat16(bf16x8 lo, bf16x8 hi) {
    return __builtin_shufflevector(lo, hi, 0, 1, 2, 3, 4, 5, 6, 7,
                                           8, 9, 10, 11, 12, 13, 14, 15);
}

__device__ __forceinline__ bf16x8 cvt8(float4 a, float4 b) {
    bf16x8 r;
    r[0] = (bf16_t)a.x; r[1] = (bf16_t)a.y; r[2] = (bf16_t)a.z; r[3] = (bf16_t)a.w;
    r[4] = (bf16_t)b.x; r[5] = (bf16_t)b.y; r[6] = (bf16_t)b.z; r[7] = (bf16_t)b.w;
    return r;
}

// Problem constants
#define BB 4
#define NN 1024
#define DD 1024
#define HH 16
#define HD 64
#define TD 768
#define TL 77

// ---------------------------------------------------------------------------
// Kernel 1: mod = silu(c) @ W_ada + b_ada        [B, 6D]
// ---------------------------------------------------------------------------
__global__ __launch_bounds__(256)
void mod_gemm_kernel(const float* __restrict__ c, const float* __restrict__ W,
                     const float* __restrict__ bias, float* __restrict__ mod) {
    const int j = blockIdx.x * 256 + threadIdx.x;     // 0 .. 6D-1
    const int b = blockIdx.y;
    const float* crow = c + (size_t)b * DD;
    float acc = 0.f;
    for (int k = 0; k < DD; ++k) {
        float cv = crow[k];
        float sv = cv / (1.f + __expf(-cv));          // SiLU
        acc = fmaf(sv, W[(size_t)k * (6 * DD) + j], acc);
    }
    mod[(size_t)b * (6 * DD) + j] = acc + bias[j];
}

// ---------------------------------------------------------------------------
// Kernel 2: h = modulate(layernorm(x), shift, scale)
// ---------------------------------------------------------------------------
__global__ __launch_bounds__(256)
void ln_mod_kernel(const float* __restrict__ x, const float* __restrict__ mod,
                   float* __restrict__ out, int chunk0) {
    const int row = blockIdx.x;           // 0 .. B*N-1
    const int b   = row / NN;
    const int tid = threadIdx.x;
    const float* xr = x + (size_t)row * DD;

    float s = 0.f, ss = 0.f;
    const float4* xr4 = (const float4*)xr;
    for (int j = tid; j < DD / 4; j += 256) {
        float4 v = xr4[j];
        s  += v.x + v.y + v.z + v.w;
        ss += v.x * v.x + v.y * v.y + v.z * v.z + v.w * v.w;
    }
    __shared__ float rs[256], rss[256];
    rs[tid] = s; rss[tid] = ss;
    __syncthreads();
    for (int st = 128; st > 0; st >>= 1) {
        if (tid < st) { rs[tid] += rs[tid + st]; rss[tid] += rss[tid + st]; }
        __syncthreads();
    }
    const float mean = rs[0] * (1.f / DD);
    const float var  = rss[0] * (1.f / DD) - mean * mean;
    const float rstd = rsqrtf(var + 1e-6f);

    const float* shift = mod + (size_t)b * (6 * DD) + (size_t)chunk0 * DD;
    const float* scale = shift + DD;
    float* orow = out + (size_t)row * DD;
    for (int j = tid; j < DD; j += 256) {
        float v = (xr[j] - mean) * rstd;
        orow[j] = v * (1.f + scale[j]) + shift[j];
    }
}

// ---------------------------------------------------------------------------
// Kernel 3: tiled bf16 WMMA GEMM with DOUBLE-BUFFERED LDS.
// C[M,N] = epi(A[M,K] @ B[K,N] + bias, res)
// Block tile 128(M) x 256(N), BK=32. 8 waves (2x4), each wave a 64x64 tile
// = 4x4 WMMA tiles -> 16 WMMAs per k-step. Software pipeline: global loads
// for tile t+1 issue before the WMMAs of tile t; converted/stored to the
// alternate LDS buffer after the MMAs; ONE barrier per k-step.
// EPI: 0 bias, 1 bias+res, 2 gelu. N%256==0, K%32==0 for all calls; M guarded.
// ---------------------------------------------------------------------------
#define BM 128
#define BN 256
#define AKP 40    // A tile K-stride in halfs (80 B, 16-B multiple)
#define BNP 264   // B tile N-stride in halfs (528 B, 16-B multiple)

template <int EPI>
__global__ __launch_bounds__(256)
void gemm_wmma_kernel(const float* __restrict__ A, const float* __restrict__ B,
                      const float* __restrict__ bias, const float* __restrict__ res,
                      float* __restrict__ C, int M, int N, int K) {
    __shared__ __align__(16) bf16_t As[2][BM][AKP];   // ping-pong  20.5 KB
    __shared__ __align__(16) bf16_t Bs[2][32][BNP];   // ping-pong  33.8 KB

    const int tid  = threadIdx.x;
    const int bm   = blockIdx.x * BM;
    const int bn   = blockIdx.y * BN;
    const int wid  = tid >> 5;
    const int lane = tid & 31;
    const int half = lane >> 4;        // 0: lanes 0-15, 1: lanes 16-31
    const int lrow = lane & 15;
    const int wm   = (wid & 1) * 64;   // wave M offset in block tile
    const int wn   = (wid >> 1) * 64;  // wave N offset in block tile

    // Staging coordinates
    const int am  = tid >> 1;             // A: row 0..127
    const int ak0 = (tid & 1) * 16;       // A: k-chunk {0,16}
    const int bk  = tid >> 5;             // B: row 0..7 (+8p)
    const int bnn = (tid & 31) * 8;       // B: n-chunk 0..248

    const int base0 = half ? 8 : 0;       // A-fragment K base (ISA 7.12.2)
    const int gm    = bm + am;
    const bool arow_ok = gm < M;

    // Per-thread staging registers (live across the WMMA section)
    float4 areg[2][2];                    // A: 2 chunks x 2 float4
    float4 breg[4][2];                    // B: 4 rows  x 2 float4

    auto load_tile = [&](int k0) {
        #pragma unroll
        for (int cidx = 0; cidx < 2; ++cidx) {
            areg[cidx][0] = make_float4(0.f, 0.f, 0.f, 0.f);
            areg[cidx][1] = make_float4(0.f, 0.f, 0.f, 0.f);
            if (arow_ok) {
                const float* ap = A + (size_t)gm * K + (k0 + ak0 + cidx * 8);
                areg[cidx][0] = *(const float4*)(ap);
                areg[cidx][1] = *(const float4*)(ap + 4);
            }
        }
        #pragma unroll
        for (int p = 0; p < 4; ++p) {
            const float* bp = B + (size_t)(k0 + bk + p * 8) * N + (bn + bnn);
            breg[p][0] = *(const float4*)(bp);
            breg[p][1] = *(const float4*)(bp + 4);
        }
    };
    auto store_tile = [&](int buf) {
        #pragma unroll
        for (int cidx = 0; cidx < 2; ++cidx)
            *(bf16x8*)&As[buf][am][ak0 + cidx * 8] = cvt8(areg[cidx][0], areg[cidx][1]);
        #pragma unroll
        for (int p = 0; p < 4; ++p)
            *(bf16x8*)&Bs[buf][bk + p * 8][bnn] = cvt8(breg[p][0], breg[p][1]);
    };

    f32x8 acc[4][4] = {};

    // Prologue: stage tile 0
    load_tile(0);
    store_tile(0);
    __syncthreads();

    const int nt = K / 32;
    for (int t = 0; t < nt; ++t) {
        const int cur = t & 1;
        // Issue global loads for tile t+1 now; waited on only after the MMAs.
        if (t + 1 < nt) load_tile((t + 1) * 32);
        if (t + 2 < nt) {   // pre-warm L2 two tiles ahead
            __builtin_prefetch(A + (size_t)gm * K + ((t + 2) * 32 + ak0), 0, 0);
            __builtin_prefetch(B + (size_t)((t + 2) * 32 + bk) * N + (bn + bnn), 0, 0);
        }

        // B fragments once per k-step (each 16 strided 16-bit LDS reads)
        bf16x16 bfr[4];
        #pragma unroll
        for (int ni = 0; ni < 4; ++ni) {
            const int nb = wn + ni * 16 + lrow;                 // lane <-> N
            #pragma unroll
            for (int s = 0; s < 16; ++s)
                bfr[ni][s] = Bs[cur][base0 + (s & 7) + (s >> 3) * 16][nb];
        }
        // A fragments per mi; 16 WMMAs per k-step
        #pragma unroll
        for (int mi = 0; mi < 4; ++mi) {
            const int m = wm + mi * 16 + lrow;                  // lane <-> M
            bf16x8 lo = *(const bf16x8*)&As[cur][m][base0];
            bf16x8 hi = *(const bf16x8*)&As[cur][m][base0 + 16];
            bf16x16 af = cat16(lo, hi);
            #pragma unroll
            for (int ni = 0; ni < 4; ++ni)
                acc[mi][ni] = wmma_bf16(af, bfr[ni], acc[mi][ni]);
        }

        // Convert + store tile t+1 into the alternate buffer, then barrier.
        if (t + 1 < nt) store_tile(cur ^ 1);
        __syncthreads();
    }

    // --- epilogue + store (C layout: lane <-> N, VGPR <-> M) ---
    #pragma unroll
    for (int mi = 0; mi < 4; ++mi) {
        #pragma unroll
        for (int ni = 0; ni < 4; ++ni) {
            const int n = bn + wn + ni * 16 + lrow;
            const float bv = bias ? bias[n] : 0.f;
            #pragma unroll
            for (int v = 0; v < 8; ++v) {
                const int m = bm + wm + mi * 16 + v + half * 8;
                if (m < M) {
                    float xv = acc[mi][ni][v] + bv;
                    if (EPI == 2) {  // tanh-approx GELU
                        float x3 = xv * xv * xv;
                        xv = 0.5f * xv *
                             (1.f + tanhf(0.7978845608028654f * (xv + 0.044715f * x3)));
                    }
                    if (EPI == 1) xv += res[(size_t)m * N + n];
                    C[(size_t)m * N + n] = xv;
                }
            }
        }
    }
}

// ---------------------------------------------------------------------------
// Kernel 4: flash attention (wave32 WMMA). One block per (b, h, 256-q block).
// Each wave owns TWO 16-row q tiles: the shared K/V fragments feed two
// independent WMMA chains (8 S-WMMAs + 8 PV-WMMAs per key tile), hiding LDS
// latency behind the matrix pipe. K staged transposed; K/V fragments: 16 scalar LDS reads; P fragments are
// 2 x ds_load_b128.
// ---------------------------------------------------------------------------
#define KKP 40   // KsT key-stride in halfs (80 B)
#define VHP 72   // Vs hd-stride in halfs (144 B)
#define PKP 40   // Ps key-stride in halfs (80 B)

__global__ __launch_bounds__(256)
void flash_attn_kernel(const float* __restrict__ Qb, const float* __restrict__ Kb,
                       const float* __restrict__ Vb, float* __restrict__ Ob,
                       int Lk, int q_stride, int kv_stride) {
    __shared__ __align__(16) bf16_t KsT[64][KKP];        // [hd][key] transposed
    __shared__ __align__(16) bf16_t Vs[32][VHP];         // [key][hd]
    __shared__ __align__(16) bf16_t Ps[8][2][16][PKP];   // per-wave/qt P [q][key]

    const int tid  = threadIdx.x;
    const int wid  = tid >> 5;
    const int lane = tid & 31;
    const int half = lane >> 4;
    const int lrow = lane & 15;
    const int base0 = half ? 8 : 0;

    const int bh = blockIdx.y;
    const int b  = bh / HH;
    const int h  = bh % HH;
    const int q0 = blockIdx.x * 256 + wid * 32;   // 2 q-tiles of 16 per wave

    // Q fragments (A layout) for both q tiles
    bf16x16 qa[2][2];                             // [qt][f]
    #pragma unroll
    for (int qt = 0; qt < 2; ++qt) {
        const float* qrow =
            Qb + ((size_t)(b * NN + q0 + qt * 16 + lrow)) * q_stride + h * HD;
        #pragma unroll
        for (int f = 0; f < 2; ++f) {
            const float* qp = qrow + f * 32 + base0;
            bf16x8 lo = cvt8(*(const float4*)(qp),      *(const float4*)(qp + 4));
            bf16x8 hi = cvt8(*(const float4*)(qp + 16), *(const float4*)(qp + 20));
            qa[qt][f] = cat16(lo, hi);
        }
    }

    float mrun[2][8], lsum[2][8];
    #pragma unroll
    for (int qt = 0; qt < 2; ++qt)
        #pragma unroll
        for (int v = 0; v < 8; ++v) { mrun[qt][v] = -1e30f; lsum[qt][v] = 0.f; }
    f32x8 o[2][4] = {};

    // staging coords: 64 hd-cols x 4 groups of 8 consecutive keys
    const int col = tid & 63;
    const int kr0 = (tid >> 6) * 8;

    const int ntile = (Lk + 31) / 32;
    for (int t = 0; t < ntile; ++t) {
        const int kt0 = t * 32;
        // --- cooperative K/V tile load; K transposed with one b128 store ---
        {
            bf16x8 kpack;
            #pragma unroll
            for (int p = 0; p < 8; ++p) {
                const int kr = kr0 + p;
                const int gk = kt0 + kr;
                float kv = 0.f, vv = 0.f;
                if (gk < Lk) {
                    const size_t roff = ((size_t)(b * Lk + gk)) * kv_stride + h * HD + col;
                    kv = Kb[roff];
                    vv = Vb[roff];
                }
                kpack[p] = (bf16_t)kv;
                Vs[kr][col] = (bf16_t)vv;
            }
            *(bf16x8*)&KsT[col][kr0] = kpack;
        }
        __syncthreads();

        // --- K fragments (shared by both q tiles), then 8 S-WMMAs ---
        bf16x16 kf[2][2];                              // [sub][f]
        #pragma unroll
        for (int sub = 0; sub < 2; ++sub)
            #pragma unroll
            for (int f = 0; f < 2; ++f) {
                #pragma unroll
                for (int s = 0; s < 16; ++s)                   // lane <-> key(N)
                    kf[sub][f][s] = KsT[f * 32 + base0 + (s & 7) + (s >> 3) * 16][sub * 16 + lrow];
            }
        f32x8 s[2][2] = {};                            // [qt][sub]
        #pragma unroll
        for (int sub = 0; sub < 2; ++sub) {
            #pragma unroll
            for (int qt = 0; qt < 2; ++qt) {
                s[qt][sub] = wmma_bf16(qa[qt][0], kf[sub][0], s[qt][sub]);
                s[qt][sub] = wmma_bf16(qa[qt][1], kf[sub][1], s[qt][sub]);
            }
            const int keyg = kt0 + sub * 16 + lrow;    // lane <-> key (C layout)
            const bool valid = keyg < Lk;
            #pragma unroll
            for (int qt = 0; qt < 2; ++qt)
                #pragma unroll
                for (int v = 0; v < 8; ++v)
                    s[qt][sub][v] = valid ? s[qt][sub][v] * 0.125f : -1e30f;
        }

        // --- online softmax per q tile ---
        #pragma unroll
        for (int qt = 0; qt < 2; ++qt) {
            #pragma unroll
            for (int v = 0; v < 8; ++v) {
                float mx = fmaxf(s[qt][0][v], s[qt][1][v]);
                mx = fmaxf(mx, __shfl_xor(mx, 1, 32));
                mx = fmaxf(mx, __shfl_xor(mx, 2, 32));
                mx = fmaxf(mx, __shfl_xor(mx, 4, 32));
                mx = fmaxf(mx, __shfl_xor(mx, 8, 32));
                const float mnew  = fmaxf(mrun[qt][v], mx);
                const float alpha = __expf(mrun[qt][v] - mnew);
                mrun[qt][v] = mnew;
                const float p0 = __expf(s[qt][0][v] - mnew);
                const float p1 = __expf(s[qt][1][v] - mnew);
                s[qt][0][v] = p0; s[qt][1][v] = p1;
                float rsum = p0 + p1;
                rsum += __shfl_xor(rsum, 1, 32);
                rsum += __shfl_xor(rsum, 2, 32);
                rsum += __shfl_xor(rsum, 4, 32);
                rsum += __shfl_xor(rsum, 8, 32);
                lsum[qt][v] = lsum[qt][v] * alpha + rsum;
                #pragma unroll
                for (int j = 0; j < 4; ++j) o[qt][j][v] *= alpha;
            }
            // P: C-layout -> per-wave LDS (wave-local, DS in-order)
            #pragma unroll
            for (int sub = 0; sub < 2; ++sub)
                #pragma unroll
                for (int v = 0; v < 8; ++v)
                    Ps[wid][qt][v + half * 8][sub * 16 + lrow] = (bf16_t)s[qt][sub][v];
        }

        // --- P A-fragments + shared V fragments, then 8 PV-WMMAs ---
        bf16x16 pa[2];
        #pragma unroll
        for (int qt = 0; qt < 2; ++qt) {
            bf16x8 plo = *(const bf16x8*)&Ps[wid][qt][lrow][base0];
            bf16x8 phi = *(const bf16x8*)&Ps[wid][qt][lrow][base0 + 16];
            pa[qt] = cat16(plo, phi);
        }
        bf16x16 vf[4];
        #pragma unroll
        for (int j = 0; j < 4; ++j) {
            #pragma unroll
            for (int s = 0; s < 16; ++s)               // lane <-> hd(N)
                vf[j][s] = Vs[base0 + (s & 7) + (s >> 3) * 16][j * 16 + lrow];
        }
        #pragma unroll
        for (int j = 0; j < 4; ++j)
            #pragma unroll
            for (int qt = 0; qt < 2; ++qt)
                o[qt][j] = wmma_bf16(pa[qt], vf[j], o[qt][j]);
        __syncthreads();
    }

    // --- normalize and store: out[b, qrow, h*64 + hd], row stride D ---
    #pragma unroll
    for (int qt = 0; qt < 2; ++qt)
        #pragma unroll
        for (int j = 0; j < 4; ++j) {
            const int hd = h * HD + j * 16 + lrow;
            #pragma unroll
            for (int v = 0; v < 8; ++v) {
                const int qr = q0 + qt * 16 + v + half * 8;
                Ob[((size_t)(b * NN + qr)) * DD + hd] = o[qt][j][v] / lsum[qt][v];
            }
        }
}

// ---------------------------------------------------------------------------
// Host launcher
// ---------------------------------------------------------------------------
extern "C" void kernel_launch(void* const* d_in, const int* in_sizes, int n_in,
                              void* d_out, int out_size, void* d_ws, size_t ws_size,
                              hipStream_t stream) {
    const float* x      = (const float*)d_in[0];
    const float* c      = (const float*)d_in[1];
    const float* text   = (const float*)d_in[2];
    const float* W_ada  = (const float*)d_in[3];
    const float* b_ada  = (const float*)d_in[4];
    const float* W_qkv  = (const float*)d_in[5];
    const float* b_qkv  = (const float*)d_in[6];
    const float* W_proj = (const float*)d_in[7];
    const float* b_proj = (const float*)d_in[8];
    const float* W_ctx  = (const float*)d_in[9];
    const float* b_ctx  = (const float*)d_in[10];
    const float* W_q    = (const float*)d_in[11];
    const float* W_k    = (const float*)d_in[12];
    const float* W_v    = (const float*)d_in[13];
    const float* W_out  = (const float*)d_in[14];
    const float* b_out  = (const float*)d_in[15];
    const float* W_fc1  = (const float*)d_in[16];
    const float* b_fc1  = (const float*)d_in[17];
    const float* W_fc2  = (const float*)d_in[18];
    const float* b_fc2  = (const float*)d_in[19];
    float* out = (float*)d_out;

    const int M  = BB * NN;       // 4096
    const int Mc = BB * TL;       // 308

    // Workspace layout (floats). hidden (4096x4096) reuses qkv+attn regions
    // (12.58M + 4.19M = exactly 16.78M floats), dead by then.
    float* ws   = (float*)d_ws;
    size_t off  = 0;
    float* modb = ws + off; off += (size_t)BB * 6 * DD;        //     24,576
    float* hbuf = ws + off; off += (size_t)M * DD;             //  4,194,304
    float* qkv  = ws + off; off += (size_t)M * 3 * DD;         // 12,582,912
    float* attn = ws + off; off += (size_t)M * DD;             //  4,194,304
    float* x1   = ws + off; off += (size_t)M * DD;             //  4,194,304
    float* x2   = ws + off; off += (size_t)M * DD;             //  4,194,304
    float* ctx  = ws + off; off += (size_t)Mc * DD;            //    315,392
    if (ws_size < off * sizeof(float)) return;                 // not enough scratch

    float* qx     = qkv;                                       // [4096,1024]
    float* kx     = qkv + (size_t)M * DD;                      // [308,1024]
    float* vx     = kx + (size_t)Mc * DD;                      // [308,1024]
    float* hidden = qkv;                                       // [4096,4096]

    dim3 blk(256);
    const int GM  = M / BM;             // 32
    const int GMc = (Mc + BM - 1) / BM; // 3

    // 1) adaLN modulation vector
    mod_gemm_kernel<<<dim3(6 * DD / 256, BB), blk, 0, stream>>>(c, W_ada, b_ada, modb);

    // 2) self-attention branch
    ln_mod_kernel<<<dim3(M), blk, 0, stream>>>(x, modb, hbuf, 0);
    gemm_wmma_kernel<0><<<dim3(GM, 3 * DD / BN), blk, 0, stream>>>(
        hbuf, W_qkv, b_qkv, nullptr, qkv, M, 3 * DD, DD);
    flash_attn_kernel<<<dim3(NN / 256, BB * HH), blk, 0, stream>>>(
        qkv, qkv + DD, qkv + 2 * DD, attn, NN, 3 * DD, 3 * DD);
    gemm_wmma_kernel<1><<<dim3(GM, DD / BN), blk, 0, stream>>>(
        attn, W_proj, b_proj, x, x1, M, DD, DD);

    // 3) cross-attention branch
    ln_mod_kernel<<<dim3(M), blk, 0, stream>>>(x1, modb, hbuf, 2);
    gemm_wmma_kernel<0><<<dim3(GMc, DD / BN), blk, 0, stream>>>(
        text, W_ctx, b_ctx, nullptr, ctx, Mc, DD, TD);
    gemm_wmma_kernel<0><<<dim3(GM, DD / BN), blk, 0, stream>>>(
        hbuf, W_q, nullptr, nullptr, qx, M, DD, DD);
    gemm_wmma_kernel<0><<<dim3(GMc, DD / BN), blk, 0, stream>>>(
        ctx, W_k, nullptr, nullptr, kx, Mc, DD, DD);
    gemm_wmma_kernel<0><<<dim3(GMc, DD / BN), blk, 0, stream>>>(
        ctx, W_v, nullptr, nullptr, vx, Mc, DD, DD);
    flash_attn_kernel<<<dim3(NN / 256, BB * HH), blk, 0, stream>>>(
        qx, kx, vx, attn, TL, DD, DD);
    gemm_wmma_kernel<1><<<dim3(GM, DD / BN), blk, 0, stream>>>(
        attn, W_out, b_out, x1, x2, M, DD, DD);

    // 4) MLP branch
    ln_mod_kernel<<<dim3(M), blk, 0, stream>>>(x2, modb, hbuf, 4);
    gemm_wmma_kernel<2><<<dim3(GM, 4 * DD / BN), blk, 0, stream>>>(
        hbuf, W_fc1, b_fc1, nullptr, hidden, M, 4 * DD, DD);
    gemm_wmma_kernel<1><<<dim3(GM, DD / BN), blk, 0, stream>>>(
        hidden, W_fc2, b_fc2, x2, out, M, DD, 4 * DD);
}
